// GraphConv_73237782331695
// MI455X (gfx1250) — matrix operand order repaired and checked by reading, not compiled, not run on hardware
//
#include <hip/hip_runtime.h>
#include <hip/hip_bf16.h>
#include <math.h>

// Problem constants (match reference)
#define NN    50000
#define EE    800000
#define DIN   64
#define HH    4
#define DOUT  16
#define DTOT  64           // H * D_OUT
#define NEG_SLOPE 0.2f

typedef __attribute__((ext_vector_type(16))) _Float16 v16h;
typedef __attribute__((ext_vector_type(8)))  float    v8f;

// ---------------------------------------------------------------------------
// float atomic max via int punning (works with -inf init)
// ---------------------------------------------------------------------------
__device__ __forceinline__ void atomicMaxF(float* addr, float val) {
    if (val >= 0.0f) {
        atomicMax((int*)addr, __float_as_int(val));
    } else {
        atomicMin((unsigned int*)addr, __float_as_uint(val));
    }
}

// ---------------------------------------------------------------------------
// WMMA fragment layouts (CDNA5 ISA 7.12.2, wave32)
//   A (16x32 f16): lane&15 = row M; (lane>>4)*8 = K base;
//     halfs 0..7 = K kb..kb+7, halfs 8..15 = K kb+16..kb+23
//   B (32x16 f16): lane&15 = col N; (lane>>4)*8 = K base; halfs as in A (kb..kb+7, kb+16..kb+23)
//   C/D (16x16 f32): VGPR j -> row j + 8*(lane>>4), col lane&15
// ---------------------------------------------------------------------------
__device__ __forceinline__ v16h load_a_frag(const float* __restrict__ row_ptr,
                                            int kt, int lane) {
    const int kb = (lane >> 4) * 8;
    const int k0 = kt * 32 + kb;
    float4 f0 = *(const float4*)(row_ptr + k0);
    float4 f1 = *(const float4*)(row_ptr + k0 + 4);
    float4 f2 = *(const float4*)(row_ptr + k0 + 16);
    float4 f3 = *(const float4*)(row_ptr + k0 + 20);
    v16h a;
    a[0]  = (_Float16)f0.x; a[1]  = (_Float16)f0.y; a[2]  = (_Float16)f0.z; a[3]  = (_Float16)f0.w;
    a[4]  = (_Float16)f1.x; a[5]  = (_Float16)f1.y; a[6]  = (_Float16)f1.z; a[7]  = (_Float16)f1.w;
    a[8]  = (_Float16)f2.x; a[9]  = (_Float16)f2.y; a[10] = (_Float16)f2.z; a[11] = (_Float16)f2.w;
    a[12] = (_Float16)f3.x; a[13] = (_Float16)f3.y; a[14] = (_Float16)f3.z; a[15] = (_Float16)f3.w;
    return a;
}

// Stage a 64x64 fp32 weight into LDS, pre-swizzled into WMMA B-fragment order:
//   wlds[(kt*4+nt)*512 + lane*16 + i] =
//       (f16) W[(kt*32 + (lane>>4)*8 + (i&7) + (i>>3)*16)*64 + nt*16 + (lane&15)]
// so a B fragment is one contiguous 32-byte per-lane read (2x ds_load_b128).
__device__ __forceinline__ void stage_weight_swizzled(const float* __restrict__ W,
                                                      _Float16* __restrict__ wlds,
                                                      int tid, int nthreads) {
    for (int idx = tid; idx < DIN * DTOT; idx += nthreads) {
        const int f    = idx >> 9;          // fragment = kt*4+nt
        const int r    = idx & 511;
        const int lane = r >> 4;
        const int i    = r & 15;
        const int kt   = f >> 2;
        const int nt   = f & 3;
        const int k    = kt * 32 + (lane >> 4) * 8 + (i & 7) + (i >> 3) * 16;
        const int n    = nt * 16 + (lane & 15);
        wlds[idx] = (_Float16)W[k * DTOT + n];
    }
}

__device__ __forceinline__ v16h load_b_frag_sw(const _Float16* __restrict__ wlds,
                                               int kt, int nt, int lane) {
    return *(const v16h*)(wlds + (size_t)(kt * 4 + nt) * 512 + lane * 16);
}

// ---------------------------------------------------------------------------
// Kernel 0: init smax = -inf, denom = 0, out = broadcast(bias)
// ---------------------------------------------------------------------------
__global__ void k_init(float* __restrict__ smax, float* __restrict__ denom,
                       float* __restrict__ out, const float* __restrict__ bias) {
    int i = blockIdx.x * blockDim.x + threadIdx.x;
    if (i < NN * HH) {
        smax[i]  = -INFINITY;
        denom[i] = 0.0f;
    }
    if (i < NN * DTOT) {
        out[i] = bias[i & (DTOT - 1)];
    }
}

// ---------------------------------------------------------------------------
// Kernel 1: x_l = node @ W_l + b_l ; x_r = node @ W_r + b_r  (WMMA f16->f32)
// Persistent waves: all 16 weight B-fragments pinned in VGPRs, grid-stride
// loop over 16-row node tiles -> no LDS traffic in steady state.
// ---------------------------------------------------------------------------
__global__ void __launch_bounds__(128)
k_node_transform(const float* __restrict__ node,
                 const float* __restrict__ Wl, const float* __restrict__ bl,
                 const float* __restrict__ Wr, const float* __restrict__ br,
                 float* __restrict__ xl, float* __restrict__ xr) {
    __shared__ alignas(32) _Float16 wl_s[DIN * DTOT];
    __shared__ alignas(32) _Float16 wr_s[DIN * DTOT];
    stage_weight_swizzled(Wl, wl_s, threadIdx.x, blockDim.x);
    stage_weight_swizzled(Wr, wr_s, threadIdx.x, blockDim.x);
    __syncthreads();

    const int lane = threadIdx.x & 31;
    const int wave = threadIdx.x >> 5;
    const int lh   = lane >> 4;
    const int nc   = lane & 15;

    // Pin all B fragments in registers (invariant across tiles).
    v16h bL[2][4], bR[2][4];
#pragma unroll
    for (int kt = 0; kt < 2; ++kt)
#pragma unroll
        for (int nt = 0; nt < 4; ++nt) {
            bL[kt][nt] = load_b_frag_sw(wl_s, kt, nt, lane);
            bR[kt][nt] = load_b_frag_sw(wr_s, kt, nt, lane);
        }

    const int ntiles = NN / 16;                          // 3125
    const int nwaves = gridDim.x * 4;
    for (int tile = blockIdx.x * 4 + wave; tile < ntiles; tile += nwaves) {
        const int row0 = tile * 16;
        const float* arow = node + (size_t)(row0 + nc) * DIN;
        v16h a0 = load_a_frag(arow, 0, lane);
        v16h a1 = load_a_frag(arow, 1, lane);

#pragma unroll
        for (int nt = 0; nt < 4; ++nt) {
            const int colg = nt * 16 + nc;
            {
                v8f acc = {};
                acc = __builtin_amdgcn_wmma_f32_16x16x32_f16(false, a0, false, bL[0][nt],
                                                             (short)0, acc, false, false);
                acc = __builtin_amdgcn_wmma_f32_16x16x32_f16(false, a1, false, bL[1][nt],
                                                             (short)0, acc, false, false);
                const float bias_c = bl[colg];
#pragma unroll
                for (int j = 0; j < 8; ++j) {
                    const int rg = row0 + j + 8 * lh;
                    xl[(size_t)rg * DTOT + colg] = acc[j] + bias_c;
                }
            }
            {
                v8f acc = {};
                acc = __builtin_amdgcn_wmma_f32_16x16x32_f16(false, a0, false, bR[0][nt],
                                                             (short)0, acc, false, false);
                acc = __builtin_amdgcn_wmma_f32_16x16x32_f16(false, a1, false, bR[1][nt],
                                                             (short)0, acc, false, false);
                const float bias_c = br[colg];
#pragma unroll
                for (int j = 0; j < 8; ++j) {
                    const int rg = row0 + j + 8 * lh;
                    xr[(size_t)rg * DTOT + colg] = acc[j] + bias_c;
                }
            }
        }
    }
}

// ---------------------------------------------------------------------------
// Kernel 2 (fused): per 16-edge tile
//   x_e = edge_feats @ W_e (WMMA, never materialized to HBM)
//   m   = leaky_relu(x_l[src] + x_r[dst] + x_e)
//   score[e,h] = <m[h,:], att[h,:]>   (16-lane shuffle reduction)
//   atomicMax smax[dst,h]
// Persistent waves: 8 W_e B-fragments pinned in VGPRs across the tile loop.
// ---------------------------------------------------------------------------
__global__ void __launch_bounds__(128)
k_edge_score(const float* __restrict__ ef, const int* __restrict__ eidx,
             const float* __restrict__ xl, const float* __restrict__ xr,
             const float* __restrict__ We, const float* __restrict__ att,
             float* __restrict__ score, float* __restrict__ smax) {
    __shared__ alignas(32) _Float16 we_s[DIN * DTOT];
    stage_weight_swizzled(We, we_s, threadIdx.x, blockDim.x);
    __syncthreads();

    const int lane = threadIdx.x & 31;
    const int wave = threadIdx.x >> 5;
    const int lh   = lane >> 4;
    const int nc   = lane & 15;

    v16h bE[2][4];
#pragma unroll
    for (int kt = 0; kt < 2; ++kt)
#pragma unroll
        for (int nt = 0; nt < 4; ++nt)
            bE[kt][nt] = load_b_frag_sw(we_s, kt, nt, lane);

    // attention vector entry for this lane, per head (invariant across tiles)
    float attv[HH];
#pragma unroll
    for (int t = 0; t < HH; ++t) attv[t] = att[t * 16 + nc];

    const int ntiles = EE / 16;                          // 50000
    const int nwaves = gridDim.x * 4;
    for (int tile = blockIdx.x * 4 + wave; tile < ntiles; tile += nwaves) {
        const int e0 = tile * 16;

        // per-lane (lane&15) source/dest of edge e0+(lane&15); broadcast via shfl
        const int sl = eidx[e0 + nc];
        const int dl = eidx[EE + e0 + nc];

        const float* arow = ef + (size_t)(e0 + nc) * DIN;
        v16h a0 = load_a_frag(arow, 0, lane);
        v16h a1 = load_a_frag(arow, 1, lane);

#pragma unroll
        for (int t = 0; t < HH; ++t) {      // one N-tile == one head (DOUT==16)
            v8f acc = {};
            acc = __builtin_amdgcn_wmma_f32_16x16x32_f16(false, a0, false, bE[0][t],
                                                         (short)0, acc, false, false);
            acc = __builtin_amdgcn_wmma_f32_16x16x32_f16(false, a1, false, bE[1][t],
                                                         (short)0, acc, false, false);
            const int col = t * 16 + nc;

#pragma unroll
            for (int j = 0; j < 8; ++j) {
                const int m = j + 8 * lh;             // edge row in tile
                const int s = __shfl(sl, m);
                const int d = __shfl(dl, m);
                float v = acc[j] + xl[(size_t)s * DTOT + col] + xr[(size_t)d * DTOT + col];
                v = (v > 0.0f) ? v : NEG_SLOPE * v;
                float w = v * attv[t];
                // sum over 16 lanes of this half (cols 0..15 of head t)
                w += __shfl_xor(w, 1);
                w += __shfl_xor(w, 2);
                w += __shfl_xor(w, 4);
                w += __shfl_xor(w, 8);
                if (nc == 0) {
                    const int e = e0 + m;
                    score[(size_t)e * HH + t] = w;
                    atomicMaxF(&smax[(size_t)d * HH + t], w);
                }
            }
        }
    }
}

// ---------------------------------------------------------------------------
// Kernel 3: ex = exp(score - smax[dst]); denom[dst] += ex (in place on score)
// ---------------------------------------------------------------------------
__global__ void k_exp_denom(const int* __restrict__ eidx,
                            float* __restrict__ score,
                            const float* __restrict__ smax,
                            float* __restrict__ denom) {
    int i = blockIdx.x * blockDim.x + threadIdx.x;
    if (i >= EE * HH) return;
    const int e = i >> 2;
    const int h = i & 3;
    const int d = eidx[EE + e];
    const float ex = expf(score[i] - smax[(size_t)d * HH + h]);
    score[i] = ex;
    atomicAdd(&denom[(size_t)d * HH + h], ex);
}

// ---------------------------------------------------------------------------
// Kernel 4: out[dst] += x_l[src] * alpha   (alpha = ex / (denom[dst]+1e-16))
// One thread handles 4 consecutive columns (one float4 of x_l, one alpha).
// ---------------------------------------------------------------------------
__global__ void k_aggregate(const int* __restrict__ eidx,
                            const float* __restrict__ xl,
                            const float* __restrict__ ex,
                            const float* __restrict__ denom,
                            float* __restrict__ out) {
    const long long i = (long long)blockIdx.x * blockDim.x + threadIdx.x;
    if (i >= (long long)EE * (DTOT / 4)) return;
    const int e  = (int)(i >> 4);          // 16 quads per edge
    const int q  = (int)(i & 15);
    const int h  = q >> 2;                 // 4 quads per head
    const int c0 = q * 4;
    const int s  = eidx[e];
    const int d  = eidx[EE + e];
    const float alpha = ex[(size_t)e * HH + h] / (denom[(size_t)d * HH + h] + 1e-16f);
    const float4 xv = *(const float4*)(xl + (size_t)s * DTOT + c0);
    float* op = out + (size_t)d * DTOT + c0;
    atomicAdd(op + 0, xv.x * alpha);
    atomicAdd(op + 1, xv.y * alpha);
    atomicAdd(op + 2, xv.z * alpha);
    atomicAdd(op + 3, xv.w * alpha);
}

// ---------------------------------------------------------------------------
// launcher
// ---------------------------------------------------------------------------
extern "C" void kernel_launch(void* const* d_in, const int* in_sizes, int n_in,
                              void* d_out, int out_size, void* d_ws, size_t ws_size,
                              hipStream_t stream) {
    const float* node = (const float*)d_in[0];
    const int*   eidx = (const int*)  d_in[1];
    const float* ef   = (const float*)d_in[2];
    const float* Wl   = (const float*)d_in[3];
    const float* bl   = (const float*)d_in[4];
    const float* Wr   = (const float*)d_in[5];
    const float* br   = (const float*)d_in[6];
    const float* We   = (const float*)d_in[7];
    const float* att  = (const float*)d_in[8];
    const float* bias = (const float*)d_in[9];
    float* out = (float*)d_out;

    float* ws    = (float*)d_ws;
    float* xl    = ws;                                   // N*64
    float* xr    = xl    + (size_t)NN * DTOT;            // N*64
    float* score = xr    + (size_t)NN * DTOT;            // E*4 (later ex)
    float* smax  = score + (size_t)EE * HH;              // N*4
    float* denom = smax  + (size_t)NN * HH;              // N*4

    // 0) init smax/denom/out
    {
        const int total = NN * DTOT;
        k_init<<<(total + 255) / 256, 256, 0, stream>>>(smax, denom, out, bias);
    }
    // 1) node transforms (WMMA), persistent waves
    {
        k_node_transform<<<256, 128, 0, stream>>>(node, Wl, bl, Wr, br, xl, xr);
    }
    // 2) fused edge GEMM + attention scores + segment max (WMMA), persistent
    {
        k_edge_score<<<1024, 128, 0, stream>>>(ef, eidx, xl, xr, We, att,
                                               score, smax);
    }
    // 3) exp + denom
    {
        const int total = EE * HH;
        k_exp_denom<<<(total + 255) / 256, 256, 0, stream>>>(eidx, score, smax, denom);
    }
    // 4) weighted scatter aggregation (4 cols / thread)
    {
        const long long total = (long long)EE * (DTOT / 4);
        const int blocks = (int)((total + 255) / 256);
        k_aggregate<<<blocks, 256, 0, stream>>>(eidx, xl, score, denom, out);
    }
}
